// Classifier_28406913696567
// MI455X (gfx1250) — matrix operand order repaired and checked by reading, not compiled, not run on hardware
//
#include <hip/hip_runtime.h>
#include <hip/hip_bf16.h>
#include <stdint.h>

#define NN 40000      // nodes
#define NE 640000     // edges
#define DIN 128
#define DHID 256
#define DOUT 256
#define NCLS 10
#define NG 128
#define NEG_SLOPE 0.01f

typedef __attribute__((ext_vector_type(16))) __bf16 v16bf;
typedef __attribute__((ext_vector_type(8)))  __bf16 v8bf;
typedef __attribute__((ext_vector_type(8)))  float  v8f;

struct bfpair { v8bf lo, hi; };   // 32 bytes, bit_cast-able to v16bf

__device__ __forceinline__ unsigned short f32_to_bf16_bits(float f) {
  unsigned int u = __builtin_bit_cast(unsigned int, f);
  u += 0x7fffu + ((u >> 16) & 1u);            // round to nearest even
  return (unsigned short)(u >> 16);
}
__device__ __forceinline__ float lrelu(float v) {
  return (v > 0.0f) ? v : v * NEG_SLOPE;
}

// ---------------- degree / normalization -----------------------------------
__global__ void k_degrees(const long long* __restrict__ src,
                          const long long* __restrict__ dst,
                          float* __restrict__ deg_out, float* __restrict__ deg_in) {
  int e = blockIdx.x * blockDim.x + threadIdx.x;
  if (e < NE) {
    atomicAdd(&deg_out[(int)src[e]], 1.0f);
    atomicAdd(&deg_in[(int)dst[e]], 1.0f);
  }
}
__global__ void k_inv_sqrt(float* __restrict__ d, int n) {
  int i = blockIdx.x * blockDim.x + threadIdx.x;
  if (i < n) d[i] = rsqrtf(fmaxf(d[i], 1.0f));
}
__global__ void k_count_nodes(const long long* __restrict__ gid, float* __restrict__ gcnt) {
  int i = blockIdx.x * blockDim.x + threadIdx.x;
  if (i < NN) atomicAdd(&gcnt[(int)gid[i]], 1.0f);
}

// ---------------- edge scatter (m = h * isq_out[src], agg[dst] += m) -------
template <int D>  // D = feature dim, multiple of 4; D/4 lanes per edge
__global__ void k_scatter(const float* __restrict__ h,
                          const long long* __restrict__ src,
                          const long long* __restrict__ dst,
                          const float* __restrict__ isq_out,
                          float* __restrict__ agg) {
  const int lanes = D / 4;
  long long gid = (long long)blockIdx.x * blockDim.x + threadIdx.x;
  if (gid >= (long long)NE * lanes) return;
  int e = (int)(gid / lanes);
  int l = (int)(gid % lanes);
  int s = (int)src[e];
  int d = (int)dst[e];
  float sc = isq_out[s];
  float4 v = ((const float4*)(h + (size_t)s * D))[l];
  float* p = agg + (size_t)d * D + l * 4;
  atomicAdd(p + 0, v.x * sc);
  atomicAdd(p + 1, v.y * sc);
  atomicAdd(p + 2, v.z * sc);
  atomicAdd(p + 3, v.w * sc);
}

// ---------------- A pre-pass: a_bf[row,k] = bf16(agg[row,k]*isq_in[row]) ---
template <int K>
__global__ void k_pack_a(const float* __restrict__ agg, const float* __restrict__ isq_in,
                         unsigned short* __restrict__ abf) {
  int t = blockIdx.x * blockDim.x + threadIdx.x;      // one float4 per thread
  if (t >= NN * K / 4) return;
  int i4 = t * 4;
  float sc = isq_in[i4 / K];
  float4 v = ((const float4*)agg)[t];
  ushort4 o;
  o.x = f32_to_bf16_bits(v.x * sc);
  o.y = f32_to_bf16_bits(v.y * sc);
  o.z = f32_to_bf16_bits(v.z * sc);
  o.w = f32_to_bf16_bits(v.w * sc);
  ((ushort4*)abf)[t] = o;
}

// ---------------- B pre-pack into WMMA lane layout -------------------------
// bpk[((kt32*16 + ntile)*32 + lane)*16 + e] = bf16(W[(kt32*32 + half*16 + e)*256
//                                                  + ntile*16 + (lane&15)])
template <int K>
__global__ void k_pack_b(const float* __restrict__ W, unsigned short* __restrict__ bpk) {
  int t = blockIdx.x * blockDim.x + threadIdx.x;
  if (t >= (K / 32) * 16 * 32) return;
  int lane  = t & 31;
  int ntile = (t >> 5) & 15;
  int kt32  = t >> 9;
  int kb    = (lane >> 4) * 16;
  int n     = ntile * 16 + (lane & 15);
  unsigned short* dst = bpk + (size_t)t * 16;
  #pragma unroll
  for (int e = 0; e < 16; ++e)
    dst[e] = f32_to_bf16_bits(W[(size_t)(kt32 * 32 + kb + e) * 256 + n]);
}

// ---------------- GraphConv GEMM: lrelu(a_bf @ W + b) via WMMA -------------
// One wave computes a 16x16 tile of the [NN, 256] output. K in {128, 256}.
// Inner loop: 2x b128 loads (A) + 2x b128 loads (B) + 1 WMMA. No VALU cvt.
// POOL=true fuses per-graph sum pooling (atomicAdd into gsum[NG,256]).
template <int K, bool POOL>
__global__ void k_gconv_gemm(const unsigned short* __restrict__ abf, // [NN,K] bf16
                             const unsigned short* __restrict__ bpk, // packed B
                             const float* __restrict__ bias,    // [256]
                             float* __restrict__ out,           // [NN,256] (if !POOL)
                             float* __restrict__ gsum,          // [NG,256] (if POOL)
                             const long long* __restrict__ gid) // [NN]     (if POOL)
{
  const int lane = threadIdx.x & 31;
  const int wave = threadIdx.x >> 5;
  const int tile = blockIdx.x * 8 + wave;     // 40000 tiles total
  const int mtile = tile >> 4;                // 2500 row tiles
  const int ntile = tile & 15;                // 16 col tiles
  const int half = lane >> 4;                 // 0 or 1
  const int l15 = lane & 15;

  const int arow = mtile * 16 + l15;          // A row owned by this lane
  const int bcol = ntile * 16 + l15;          // B col owned by this lane
  const int kb_a = half * 16;                 // A: K runs {kb..kb+15}

  const __bf16* aptr = (const __bf16*)abf + (size_t)arow * K + kb_a;
  const __bf16* bptr = (const __bf16*)bpk + ((size_t)ntile * 32 + lane) * 16;

  v8f c = {};
  #pragma unroll
  for (int kt = 0; kt < K; kt += 32) {
    const v8bf* ap = (const v8bf*)(aptr + kt);          // 16B aligned
    bfpair pa{ap[0], ap[1]};                            // k-run +0..15
    const v8bf* bp = (const v8bf*)(bptr + (size_t)(kt >> 5) * (16 * 32 * 16));
    bfpair pb{bp[0], bp[1]};                            // 16 contiguous bf16
    v16bf a = __builtin_bit_cast(v16bf, pa);
    v16bf b = __builtin_bit_cast(v16bf, pb);
    c = __builtin_amdgcn_wmma_f32_16x16x32_bf16(
        /*neg_a=*/false, a, /*neg_b=*/false, b,
        /*c_mod=*/(short)0, c, /*reuse_a=*/false, /*reuse_b=*/false);
  }

  const float bn = bias[bcol];
  const int moff = half * 8;                  // C/D: VGPR e -> row e + moff
  if (POOL) {
    #pragma unroll
    for (int e = 0; e < 8; ++e) {
      int row = mtile * 16 + moff + e;
      float v = lrelu(c[e] + bn);
      atomicAdd(&gsum[(size_t)((int)gid[row]) * 256 + bcol], v);
    }
  } else {
    #pragma unroll
    for (int e = 0; e < 8; ++e) {
      int row = mtile * 16 + moff + e;
      out[(size_t)row * 256 + bcol] = lrelu(c[e] + bn);
    }
  }
}

// ---------------- final head: (gsum/cnt) @ Wc + bc -------------------------
__global__ void k_classifier(const float* __restrict__ gsum, const float* __restrict__ gcnt,
                             const float* __restrict__ Wc, const float* __restrict__ bc,
                             float* __restrict__ out) {
  int t = blockIdx.x * blockDim.x + threadIdx.x;
  if (t >= NG * NCLS) return;
  int g = t / NCLS, cl = t % NCLS;
  float inv = 1.0f / fmaxf(gcnt[g], 1.0f);
  float acc = 0.0f;
  for (int k = 0; k < DOUT; ++k) acc += gsum[(size_t)g * DOUT + k] * Wc[k * NCLS + cl];
  out[t] = acc * inv + bc[cl];
}

extern "C" void kernel_launch(void* const* d_in, const int* in_sizes, int n_in,
                              void* d_out, int out_size, void* d_ws, size_t ws_size,
                              hipStream_t stream) {
  const float*     x    = (const float*)d_in[0];
  const long long* src  = (const long long*)d_in[1];
  const long long* dst  = (const long long*)d_in[2];
  const long long* gids = (const long long*)d_in[3];
  const float*     W1   = (const float*)d_in[4];
  const float*     b1   = (const float*)d_in[5];
  const float*     W2   = (const float*)d_in[6];
  const float*     b2   = (const float*)d_in[7];
  const float*     Wc   = (const float*)d_in[8];
  const float*     bc   = (const float*)d_in[9];
  float* out = (float*)d_out;

  // workspace carve-up (all region sizes multiples of 1024 floats -> 4KB align)
  float* ws = (float*)d_ws;
  float* deg_out = ws; ws += 40960;                   // reused as isq_out
  float* deg_in  = ws; ws += 40960;                   // reused as isq_in
  float* gsum    = ws; ws += NG * DOUT;               // 32768
  float* gcnt    = ws; ws += 1024;
  float* agg1    = ws; ws += (size_t)NN * DIN;        // 5.12M f
  float* h1      = ws; ws += (size_t)NN * DHID;       // 10.24M f
  float* agg2    = ws; ws += (size_t)NN * DHID;       // 10.24M f
  unsigned short* a1bf = (unsigned short*)ws; ws += (size_t)NN * DIN  / 2;
  unsigned short* a2bf = (unsigned short*)ws; ws += (size_t)NN * DHID / 2;
  unsigned short* b1pk = (unsigned short*)ws; ws += (DIN  / 32) * 16 * 32 * 16 / 2;
  unsigned short* b2pk = (unsigned short*)ws; ws += (DHID / 32) * 16 * 32 * 16 / 2;

  // zero accumulators (deg_out..gcnt are contiguous)
  hipMemsetAsync(deg_out, 0, (size_t)(40960 + 40960 + NG * DOUT + 1024) * sizeof(float), stream);
  hipMemsetAsync(agg1, 0, (size_t)NN * DIN * sizeof(float), stream);
  hipMemsetAsync(agg2, 0, (size_t)NN * DHID * sizeof(float), stream);

  // degrees -> inv-sqrt normalizers
  k_degrees<<<(NE + 255) / 256, 256, 0, stream>>>(src, dst, deg_out, deg_in);
  k_inv_sqrt<<<(NN + 255) / 256, 256, 0, stream>>>(deg_out, NN);
  k_inv_sqrt<<<(NN + 255) / 256, 256, 0, stream>>>(deg_in, NN);

  // weights -> packed bf16 WMMA-B layout; per-graph node counts
  k_pack_b<DIN ><<<((DIN  / 32) * 16 * 32 + 255) / 256, 256, 0, stream>>>(W1, b1pk);
  k_pack_b<DHID><<<((DHID / 32) * 16 * 32 + 255) / 256, 256, 0, stream>>>(W2, b2pk);
  k_count_nodes<<<(NN + 255) / 256, 256, 0, stream>>>(gids, gcnt);

  const int gemm_blocks = (NN / 16) * (256 / 16) / 8;  // 5000 blocks x 8 waves

  // layer 1: scatter -> scale+cvt -> WMMA gemm (+bias, lrelu) -> h1
  k_scatter<DIN><<<(int)(((long long)NE * (DIN / 4) + 255) / 256), 256, 0, stream>>>(
      x, src, dst, deg_out, agg1);
  k_pack_a<DIN><<<(NN * DIN / 4 + 255) / 256, 256, 0, stream>>>(agg1, deg_in, a1bf);
  k_gconv_gemm<DIN, false><<<gemm_blocks, 256, 0, stream>>>(
      a1bf, b1pk, b1, h1, nullptr, nullptr);

  // layer 2: scatter -> scale+cvt -> WMMA gemm fused with graph-sum pooling
  k_scatter<DHID><<<(int)(((long long)NE * (DHID / 4) + 255) / 256), 256, 0, stream>>>(
      h1, src, dst, deg_out, agg2);
  k_pack_a<DHID><<<(NN * DHID / 4 + 255) / 256, 256, 0, stream>>>(agg2, deg_in, a2bf);
  k_gconv_gemm<DHID, true><<<gemm_blocks, 256, 0, stream>>>(
      a2bf, b2pk, b2, nullptr, gsum, gids);

  // head
  k_classifier<<<(NG * NCLS + 255) / 256, 256, 0, stream>>>(gsum, gcnt, Wc, bc, out);
}
